// DensityEncoder_51565377356250
// MI455X (gfx1250) — compile-verified, matrix operand order checked by reading, not run on hardware
//
#include <hip/hip_runtime.h>
#include <hip/hip_bf16.h>
#include <stdint.h>

// ---------------------------------------------------------------------------
// Problem constants (from reference)
// ---------------------------------------------------------------------------
#define NN 8192
#define KK 30
#define EE (NN * KK)      // 245760 edges, divisible by 256
#define CZ 128

typedef _Float16 h16;
typedef __attribute__((ext_vector_type(16))) _Float16 v16h;
typedef __attribute__((ext_vector_type(8)))  float    v8f;
typedef __attribute__((ext_vector_type(4)))  unsigned int v4u;
typedef __attribute__((ext_vector_type(8)))  int      v8i;
typedef __attribute__((ext_vector_type(4)))  int      v4i;

#if defined(__has_builtin)
#if __has_builtin(__builtin_amdgcn_tensor_load_to_lds) && \
    __has_builtin(__builtin_amdgcn_s_wait_tensorcnt)
#define HAVE_TDM 1
#endif
#endif

// ---------------------------------------------------------------------------
// WMMA 16-bit fragment element mapping (ISA 7.12.2).
// Fragment element i of lane (hs*16 + r):
//   i in [0,8)  -> k = 32*t + hs*8 + i        (contiguous run of 8)
//   i in [8,16) -> k = 32*t + 16 + hs*8 + i-8 (contiguous run of 8)
// a_swz(): flat offset of (row m, k) in a fragment-order buffer:
//   offset = ((mt*KT + t)*32 + lane)*16 + i
// ---------------------------------------------------------------------------
__device__ __forceinline__ size_t a_swz(size_t m, int k, int Kp) {
  size_t mt = m >> 4;
  int r = (int)(m & 15);
  int t = k >> 5, kl = k & 31;
  int hs, i;
  if ((kl & 15) < 8) { hs = 0; } else { hs = 1; }
  i = (kl & 7) + ((kl >= 16) ? 8 : 0);
  return ((mt * (size_t)(Kp >> 5) + (size_t)t) * 32 + (size_t)(hs * 16 + r)) * 16 +
         (size_t)i;
}

// ---------------------------------------------------------------------------
// GEMM: out = act(A[M x Kp] * W[Kp x Np] + bias), operands in fragment order.
// No LDS: A streams from HBM (zero cross-wave reuse), W fragments live in the
// 192MB L2. Wave owns 2 M-tiles x NTILES N-tiles. NTILES is compile-time so
// the K-loop is a branch-free clause of b128 loads + v_wmma, and the epilogue
// is fully unrolled (no dynamic acc indexing -> no scratch spills).
// f16 output is written back in fragment order (stride Np) for the next GEMM;
// f32 output is row-major with true width Nt.
// ---------------------------------------------------------------------------
#define MW 2  // row-tiles per wave; block = 8 waves -> 256 rows

template <int NTILES>
__global__ __launch_bounds__(256) void gemm_f16_wmma(
    const h16* __restrict__ A, const h16* __restrict__ W,
    const float* __restrict__ bias, int M, int Kp, int Np, int Nt,
    h16* __restrict__ outH, float* __restrict__ outF, int act) {
  const int lane = threadIdx.x & 31;
  const int wave = threadIdx.x >> 5;
  const int KT = Kp >> 5;                       // K tiles of 32
  const int NT = Np >> 4;                       // N tiles of 16
  const size_t mt0 = (size_t)blockIdx.x * 16 + (size_t)(wave * MW);
  const int jBase = blockIdx.y * NTILES;

  const v16h* __restrict__ Af = (const v16h*)A;
  const v16h* __restrict__ Wf = (const v16h*)W;

  v8f acc[MW][NTILES];
#pragma unroll
  for (int mw = 0; mw < MW; ++mw)
#pragma unroll
    for (int j = 0; j < NTILES; ++j)
      acc[mw][j] = v8f{0.f, 0.f, 0.f, 0.f, 0.f, 0.f, 0.f, 0.f};

  for (int t = 0; t < KT; ++t) {
    v16h a0 = Af[(mt0 * (size_t)KT + t) * 32 + lane];
    v16h a1 = Af[((mt0 + 1) * (size_t)KT + t) * 32 + lane];
    // speculative prefetch of next A chunk (harmlessly dropped past the end)
    __builtin_prefetch(&Af[(mt0 * (size_t)KT + t + 1) * 32 + lane], 0, 1);
#pragma unroll
    for (int j = 0; j < NTILES; ++j) {
      v16h b = Wf[((size_t)t * NT + jBase + j) * 32 + lane];
      acc[0][j] = __builtin_amdgcn_wmma_f32_16x16x32_f16(
          false, a0, false, b, (short)0, acc[0][j], false, false);
      acc[1][j] = __builtin_amdgcn_wmma_f32_16x16x32_f16(
          false, a1, false, b, (short)0, acc[1][j], false, false);
    }
  }

  // C/D layout: lane -> n = lane&15, VGPR r -> m = r + 8*(lane>>4)
  const int nl = lane & 15, hs = lane >> 4;
  const bool hasH = outH != nullptr, hasF = outF != nullptr;
#pragma unroll
  for (int mw = 0; mw < MW; ++mw) {
#pragma unroll
    for (int j = 0; j < NTILES; ++j) {
      int n = (jBase + j) * 16 + nl;
      if (n >= Nt) continue;
      float bv = bias ? bias[n] : 0.f;
#pragma unroll
      for (int r = 0; r < 8; ++r) {
        size_t m = (mt0 + mw) * 16 + (size_t)(hs * 8 + r);
        float v = acc[mw][j][r] + bv;
        if (act == 1) v = v > 0.f ? v : 0.f;
        else if (act == 2) v = 1.f / (1.f + __expf(-v));
        if (hasH) outH[a_swz(m, n, Np)] = (h16)v;   // fragment order for next GEMM
        if (hasF) outF[m * (size_t)Nt + (size_t)n] = v;
      }
    }
  }
  (void)M;
}

// ---------------------------------------------------------------------------
// Tensor Data Mover: 1-D tile of f32 from global -> LDS (D# per ISA 8.3/8.4)
// ---------------------------------------------------------------------------
#if HAVE_TDM
__device__ __forceinline__ void tdm_load_1d_f32(uint32_t lds_off, const void* gsrc,
                                                uint32_t total_elems,
                                                uint32_t tile_elems) {
  uint64_t ga = (uint64_t)(uintptr_t)gsrc;
  v4u g0;
  g0[0] = 1u;                                    // count=1, user descriptor
  g0[1] = lds_off;                               // lds_addr (bytes)
  g0[2] = (uint32_t)(ga & 0xFFFFFFFFu);          // global_addr[31:0]
  g0[3] = (uint32_t)((ga >> 32) & 0x01FFFFFFu)   // global_addr[56:32]
          | 0x80000000u;                         // type=2 ("image")
  v8i g1;
  g1[0] = 0x00020000;                            // data_size=2 (4 bytes)
  g1[1] = (int)((total_elems & 0xFFFFu) << 16);  // tensor_dim0[15:0] @ bit48
  g1[2] = (int)((total_elems >> 16) & 0xFFFFu);  // tensor_dim0[31:16]
  g1[3] = (int)((tile_elems & 0xFFFFu) << 16);   // tile_dim0 @ bit112
  g1[4] = 0;                                     // tile_dim1=0, tile_dim2=0
  g1[5] = (int)total_elems;                      // tensor_dim0_stride[31:0]
  g1[6] = 0;
  g1[7] = 0;
  v4i z4 = {0, 0, 0, 0};
#if __clang_major__ >= 23
  v8i z8 = {0, 0, 0, 0, 0, 0, 0, 0};
  __builtin_amdgcn_tensor_load_to_lds(g0, g1, z4, z4, z8, 0);
#else
  __builtin_amdgcn_tensor_load_to_lds(g0, g1, z4, z4, 0);
#endif
}
#endif

// ---------------------------------------------------------------------------
// kNN: one thread per node, LDS-tiled candidate scan (tiles DMA'd by the TDM),
// sorted top-30 insertion (matches top_k(-d2,K): ascending d2; dst=i, src=nbr)
// ---------------------------------------------------------------------------
__global__ __launch_bounds__(256) void knn_kernel(
    const float* __restrict__ X, const float* __restrict__ mask,
    int* __restrict__ src, int* __restrict__ dst) {
  __shared__ float sx[256 * 3];
  const int tid = threadIdx.x;
  const int i = blockIdx.x * 256 + tid;
  const float xi0 = X[i * 3 + 0], xi1 = X[i * 3 + 1], xi2 = X[i * 3 + 2];
  float bd[KK];
  int   bi[KK];
#pragma unroll
  for (int k = 0; k < KK; ++k) { bd[k] = 3.0e38f; bi[k] = k; }

#if HAVE_TDM
  const uint32_t sx_off = (uint32_t)(size_t)(__attribute__((address_space(3))) char*)(char*)sx;
#endif

  for (int jt = 0; jt < NN / 256; ++jt) {
    __syncthreads();  // previous tile fully consumed
#if HAVE_TDM
    if (tid < 32) {   // one wave issues the DMA (TDM ignores EXEC)
      tdm_load_1d_f32(sx_off, X + (size_t)jt * 768, NN * 3, 768);
      __builtin_amdgcn_s_wait_tensorcnt(0);
    }
    __syncthreads();
    {  // apply residue mask on the staged tile
      int j = jt * 256 + tid;
      if (!(mask[j] > 0.f)) {
        sx[tid * 3 + 0] = 1.0e18f; sx[tid * 3 + 1] = 1.0e18f; sx[tid * 3 + 2] = 1.0e18f;
      }
    }
    __syncthreads();
#else
    {
      int j = jt * 256 + tid;
      float m = mask[j];
      sx[tid * 3 + 0] = (m > 0.f) ? X[j * 3 + 0] : 1.0e18f;
      sx[tid * 3 + 1] = (m > 0.f) ? X[j * 3 + 1] : 1.0e18f;
      sx[tid * 3 + 2] = (m > 0.f) ? X[j * 3 + 2] : 1.0e18f;
    }
    __syncthreads();
#endif
    for (int jj = 0; jj < 256; ++jj) {
      int jn = jt * 256 + jj;
      if (jn == i) continue;
      float dx = sx[jj * 3 + 0] - xi0;
      float dy = sx[jj * 3 + 1] - xi1;
      float dz = sx[jj * 3 + 2] - xi2;
      float d2 = dx * dx + dy * dy + dz * dz;
      if (d2 < bd[KK - 1]) {
        int p = KK - 1;
        while (p > 0 && bd[p - 1] > d2) {
          bd[p] = bd[p - 1]; bi[p] = bi[p - 1]; --p;
        }
        bd[p] = d2; bi[p] = jn;
      }
    }
  }
  for (int k = 0; k < KK; ++k) {
    src[i * KK + k] = bi[k];
    dst[i * KK + k] = i;
  }
}

// ---------------------------------------------------------------------------
// Virtual CB + bb5 (N x 5 x 3)
// ---------------------------------------------------------------------------
__global__ __launch_bounds__(256) void bb5_kernel(
    const float* __restrict__ bb, float* __restrict__ bb5) {
  int n = blockIdx.x * 256 + threadIdx.x;
  if (n >= NN) return;
  float a4[4][3];
#pragma unroll
  for (int a = 0; a < 4; ++a)
#pragma unroll
    for (int c = 0; c < 3; ++c) {
      a4[a][c] = bb[(size_t)n * 12 + a * 3 + c];
      bb5[(size_t)n * 15 + a * 3 + c] = a4[a][c];
    }
  float b0 = a4[1][0] - a4[0][0], b1 = a4[1][1] - a4[0][1], b2 = a4[1][2] - a4[0][2];
  float c0 = a4[2][0] - a4[1][0], c1 = a4[2][1] - a4[1][1], c2 = a4[2][2] - a4[1][2];
  float x0 = b1 * c2 - b2 * c1, x1 = b2 * c0 - b0 * c2, x2 = b0 * c1 - b1 * c0;
  bb5[(size_t)n * 15 + 12] = -0.58273431f * x0 + 0.56802827f * b0 - 0.54067466f * c0 + a4[1][0];
  bb5[(size_t)n * 15 + 13] = -0.58273431f * x1 + 0.56802827f * b1 - 0.54067466f * c1 + a4[1][1];
  bb5[(size_t)n * 15 + 14] = -0.58273431f * x2 + 0.56802827f * b2 - 0.54067466f * c2 + a4[1][2];
}

// ---------------------------------------------------------------------------
// Edge features, fragment-vectorized: one thread per fragment-lane builds a
// whole v16h (two 8-runs, each inside one RBF pair) and stores it with b128s.
// Feature order: 25 pairs x 16 RBF (400) then 16 positional.  Kp = 416.
// ---------------------------------------------------------------------------
__global__ __launch_bounds__(256) void edge_feat_kernel(
    const float* __restrict__ bb5, const int* __restrict__ src,
    const int* __restrict__ dst, h16* __restrict__ out) {
  const int KT = 13;  // 416/32
  size_t idx = (size_t)blockIdx.x * 256 + threadIdx.x;
  size_t total = (size_t)(EE / 16) * KT * 32;
  if (idx >= total) return;
  const int lane = (int)(idx & 31);
  size_t rest = idx >> 5;
  const int t = (int)(rest % KT);
  const size_t mt = rest / KT;
  const int r = lane & 15, hs = lane >> 4;
  const size_t e = mt * 16 + r;
  const int s = src[e], d = dst[e];
  const float rel = (float)(d - s);
  v16h frag;
#pragma unroll
  for (int h = 0; h < 2; ++h) {
    const int kb = t * 32 + h * 16 + hs * 8;   // 8-aligned run start
    float dist = 0.f;
    if (kb < 400) {  // run lies wholly inside RBF pair kb>>4
      int pair = kb >> 4;
      int a = pair / 5, b = pair % 5;
      float dx = bb5[(size_t)s * 15 + a * 3 + 0] - bb5[(size_t)d * 15 + b * 3 + 0] + 1e-8f;
      float dy = bb5[(size_t)s * 15 + a * 3 + 1] - bb5[(size_t)d * 15 + b * 3 + 1] + 1e-8f;
      float dz = bb5[(size_t)s * 15 + a * 3 + 2] - bb5[(size_t)d * 15 + b * 3 + 2] + 1e-8f;
      dist = sqrtf(dx * dx + dy * dy + dz * dz);
    }
#pragma unroll
    for (int i = 0; i < 8; ++i) {
      int f = kb + i;
      float v;
      if (f < 400) {
        int rr = f & 15;
        float mu = 2.0f + (float)rr * (20.0f / 15.0f);  // linspace(2,22,16)
        float tt = (dist - mu) * (1.0f / 1.25f);        // sigma = 20/16
        v = __expf(-tt * tt);
      } else {
        int r2 = f - 400, j = r2 & 7;
        float freq = __expf((2.0f * (float)j) * (-logf(10000.0f) / 16.0f));
        float ang = rel * freq;
        v = (r2 < 8) ? __cosf(ang) : __sinf(ang);
      }
      frag[h * 8 + i] = (h16)v;
    }
  }
  ((v16h*)out)[idx] = frag;
}

// ---------------------------------------------------------------------------
// Weight f32 -> f16, zero-padded to (Kp, Np), stored in B-fragment order
// ---------------------------------------------------------------------------
__global__ __launch_bounds__(256) void convert_w_kernel(
    const float* __restrict__ W, h16* __restrict__ out, int K, int Nd, int Kp, int Np) {
  int idx = blockIdx.x * 256 + threadIdx.x;
  if (idx >= Kp * Np) return;
  int i = idx & 15;
  int lane = (idx >> 4) & 31;
  int rest = idx >> 9;
  int NT = Np >> 4;
  int j = rest % NT, t = rest / NT;
  int hs = lane >> 4;
  int kl = (i < 8) ? (hs * 8 + i) : (16 + hs * 8 + (i - 8));
  int k = t * 32 + kl;
  int n = j * 16 + (lane & 15);
  out[idx] = (k < K && n < Nd) ? (h16)W[(size_t)k * Nd + n] : (h16)0.0f;
}

// ---------------------------------------------------------------------------
// Fragment-vectorized concat: builds [nf[src,0,:ci], nf[dst,0,:ci], ef32]
// zero-padded to Kp, one v16h per thread.  ci==0 -> pure ef32 f32->f16.
// ---------------------------------------------------------------------------
__global__ __launch_bounds__(256) void concat_kernel(
    const float* __restrict__ nf, int ci, const int* __restrict__ src,
    const int* __restrict__ dst, const float* __restrict__ ef32,
    h16* __restrict__ out, int Kp) {
  size_t idx = (size_t)blockIdx.x * 256 + threadIdx.x;
  const int KT = Kp >> 5;
  size_t total = (size_t)(EE / 16) * KT * 32;
  if (idx >= total) return;
  const int lane = (int)(idx & 31);
  size_t rest = idx >> 5;
  const int t = (int)(rest % KT);
  const size_t mt = rest / KT;
  const int r = lane & 15, hs = lane >> 4;
  const size_t e = mt * 16 + r;
  int s = 0, d = 0;
  if (ci > 0) { s = src[e]; d = dst[e]; }
  v16h frag;
#pragma unroll
  for (int h = 0; h < 2; ++h) {
    const int kb = t * 32 + h * 16 + hs * 8;
#pragma unroll
    for (int i = 0; i < 8; ++i) {
      int c = kb + i;
      float v = 0.f;
      if (ci == 0) {
        if (c < CZ) v = ef32[e * CZ + c];
      } else if (c < ci) {
        v = nf[(size_t)s * 4 * ci + c];
      } else if (c < 2 * ci) {
        v = nf[(size_t)d * 4 * ci + (c - ci)];
      } else if (c < 2 * ci + CZ) {
        v = ef32[e * CZ + (c - 2 * ci)];
      }
      frag[h * 8 + i] = (h16)v;
    }
  }
  ((v16h*)out)[idx] = frag;
}

__global__ __launch_bounds__(256) void fill0_kernel(float* __restrict__ p, int n) {
  int i = blockIdx.x * 256 + threadIdx.x;
  if (i < n) p[i] = 0.f;
}

// segment_sum(node_feat[src] * gate) : atomics into nf (N x 4 x 32)
__global__ __launch_bounds__(256) void scatter0_kernel(
    const float* __restrict__ node_feat, const float* __restrict__ gate,
    const int* __restrict__ src, const int* __restrict__ dst,
    float* __restrict__ nf) {
  size_t idx = (size_t)blockIdx.x * 256 + threadIdx.x;
  if (idx >= (size_t)EE * 128) return;
  size_t e = idx >> 7;
  int lc = (int)(idx & 127);
  int c = lc & 31;
  float v = node_feat[(size_t)src[e] * 128 + lc] * gate[e * 32 + c];
  atomicAdd(&nf[(size_t)dst[e] * 128 + lc], v);
}

// segment_sum( (nf_old[src] @ Wval) * gate ) : nf_new (N x 4 x co)
__global__ __launch_bounds__(256) void scatter_layer_kernel(
    const float* __restrict__ nfOld, int ci, const float* __restrict__ Wval, int co,
    const float* __restrict__ gate, const int* __restrict__ src,
    const int* __restrict__ dst, float* __restrict__ nfNew) {
  size_t idx = (size_t)blockIdx.x * 256 + threadIdx.x;
  size_t total = (size_t)EE * 4 * (size_t)co;
  if (idx >= total) return;
  size_t e = idx / (size_t)(4 * co);
  int rem = (int)(idx % (size_t)(4 * co));
  int l = rem / co, d = rem % co;
  const float* row = nfOld + (size_t)src[e] * 4 * ci + (size_t)l * ci;
  float acc = 0.f;
  for (int c = 0; c < ci; ++c) acc += row[c] * Wval[(size_t)c * co + d];
  atomicAdd(&nfNew[(size_t)dst[e] * 4 * co + (size_t)l * co + d],
            acc * gate[e * (size_t)co + d]);
}

__global__ __launch_bounds__(256) void mask_mul_kernel(
    float* __restrict__ nf, const float* __restrict__ mask, int W) {
  size_t idx = (size_t)blockIdx.x * 256 + threadIdx.x;
  if (idx >= (size_t)NN * (size_t)W) return;
  nf[idx] *= mask[idx / (size_t)W];
}

// ---------------------------------------------------------------------------
// LayerNorm over 128 cols, one wave per row, optional residual input
// ---------------------------------------------------------------------------
__global__ __launch_bounds__(256) void layernorm_kernel(
    const float* __restrict__ x, const float* __restrict__ res,
    const float* __restrict__ g, const float* __restrict__ b,
    float* __restrict__ out, int rows) {
  const int wave = threadIdx.x >> 5, lane = threadIdx.x & 31;
  const size_t row = (size_t)blockIdx.x * 8 + wave;
  if (row >= (size_t)rows) return;
  float v[4];
  float s = 0.f;
#pragma unroll
  for (int i = 0; i < 4; ++i) {
    int c = i * 32 + lane;
    v[i] = x[row * CZ + c] + (res ? res[row * CZ + c] : 0.f);
    s += v[i];
  }
#pragma unroll
  for (int m = 16; m >= 1; m >>= 1) s += __shfl_xor(s, m);
  float mu = s * (1.0f / 128.0f);
  float q = 0.f;
#pragma unroll
  for (int i = 0; i < 4; ++i) { float d = v[i] - mu; q += d * d; }
#pragma unroll
  for (int m = 16; m >= 1; m >>= 1) q += __shfl_xor(q, m);
  float inv = rsqrtf(q * (1.0f / 128.0f) + 1e-5f);
#pragma unroll
  for (int i = 0; i < 4; ++i) {
    int c = i * 32 + lane;
    out[row * CZ + c] = (v[i] - mu) * inv * g[c] + b[c];
  }
}

// ---------------------------------------------------------------------------
// Final heads: rms norms, mags, rotation, 40->32->(mu,logvar)
// ---------------------------------------------------------------------------
__global__ __launch_bounds__(256) void final_kernel(
    const float* __restrict__ nf, const float* __restrict__ rot,
    const float* __restrict__ mask, const float* __restrict__ ng,
    const float* __restrict__ Wf, const float* __restrict__ bf,
    const float* __restrict__ Wmu, const float* __restrict__ bmu,
    const float* __restrict__ Wlv, const float* __restrict__ blv,
    float* __restrict__ out) {
  int n = blockIdx.x * 256 + threadIdx.x;
  if (n >= NN) return;
  const float* p = nf + (size_t)n * 32;  // (4 x 8)
  float feats[40];
  float vv[3][8];
#pragma unroll
  for (int c = 0; c < 8; ++c) {
    float x = p[c];
    feats[c] = x * rsqrtf(x * x + 1e-8f) * ng[c];
  }
#pragma unroll
  for (int c = 0; c < 8; ++c) {
    float a = p[8 + c], b = p[16 + c], d = p[24 + c];
    float m = (a * a + b * b + d * d) * (1.0f / 3.0f);
    float sc = rsqrtf(m + 1e-8f) * ng[8 + c];
    vv[0][c] = a * sc; vv[1][c] = b * sc; vv[2][c] = d * sc;
  }
#pragma unroll
  for (int c = 0; c < 8; ++c) {
    float a = vv[0][c] + 1e-8f, b = vv[1][c] + 1e-8f, d = vv[2][c] + 1e-8f;
    feats[8 + c] = sqrtf(a * a + b * b + d * d);
  }
  const float* R = rot + (size_t)n * 9;
#pragma unroll
  for (int c = 0; c < 8; ++c)
#pragma unroll
    for (int i = 0; i < 3; ++i)
      feats[16 + c * 3 + i] =
          R[i * 3 + 0] * vv[0][c] + R[i * 3 + 1] * vv[1][c] + R[i * 3 + 2] * vv[2][c];
  float mk = mask[n];
  float f2[32];
  for (int o = 0; o < 32; ++o) {
    float a = bf[o];
    for (int k = 0; k < 40; ++k) a += feats[k] * Wf[k * 32 + o];
    f2[o] = a * mk;
  }
  for (int o = 0; o < 32; ++o) {
    float mu = bmu[o], lv = blv[o];
    for (int k = 0; k < 32; ++k) {
      mu += f2[k] * Wmu[k * 32 + o];
      lv += f2[k] * Wlv[k * 32 + o];
    }
    out[(size_t)n * 32 + o] = mu;
    out[(size_t)NN * 32 + (size_t)n * 32 + o] = lv;
  }
}

// ---------------------------------------------------------------------------
// Host orchestration
// ---------------------------------------------------------------------------
extern "C" void kernel_launch(void* const* d_in, const int* in_sizes, int n_in,
                              void* d_out, int out_size, void* d_ws, size_t ws_size,
                              hipStream_t stream) {
  const float* X    = (const float*)d_in[0];
  const float* bb   = (const float*)d_in[1];
  const float* node = (const float*)d_in[2];
  const float* mask = (const float*)d_in[3];
  const float* rot  = (const float*)d_in[4];
  auto P = [&](int i) -> const float* { return (const float*)d_in[5 + i]; };
  // param leaves in setup_inputs()/dict-insertion order
  const float *We1 = P(0), *be1 = P(1), *We2 = P(2), *be2 = P(3), *We3 = P(4),
              *be3 = P(5), *ge = P(6), *bge = P(7), *Wg = P(8), *bg = P(9);
  const float *Wg0 = P(10), *bg0 = P(11), *Wv0 = P(12), *Weu1_0 = P(13),
              *beu1_0 = P(14), *Weu2_0 = P(15), *beu2_0 = P(16), *geu0 = P(17),
              *bgeu0 = P(18);
  const float *Wg1 = P(19), *bg1 = P(20), *Wv1 = P(21);
  const float *normg = P(28), *Wf = P(29), *bf = P(30), *Wmu = P(31),
              *bmu = P(32), *Wlv = P(33), *blv = P(34);

  // workspace carve (~470 MB peak; gate and eu_out alias the big arenas)
  char* ws = (char*)d_ws;
  size_t off = 0;
  auto carve = [&](size_t bytes) -> char* {
    char* p = ws + off;
    off += (bytes + 255) & ~(size_t)255;
    return p;
  };
  int*   src  = (int*)carve((size_t)EE * 4);
  int*   dst  = (int*)carve((size_t)EE * 4);
  float* bb5  = (float*)carve((size_t)NN * 15 * 4);
  h16* hWe1   = (h16*)carve((size_t)416 * 256 * 2);
  h16* hWe2   = (h16*)carve((size_t)256 * 256 * 2);
  h16* hWe3   = (h16*)carve((size_t)256 * 128 * 2);
  h16* hWg    = (h16*)carve((size_t)128 * 32 * 2);
  h16* hWg0   = (h16*)carve((size_t)192 * 16 * 2);
  h16* hWeu10 = (h16*)carve((size_t)160 * 256 * 2);
  h16* hWeu20 = (h16*)carve((size_t)256 * 128 * 2);
  h16* hWg1   = (h16*)carve((size_t)160 * 16 * 2);   // N padded 8->16
  h16* Abuf   = (h16*)carve((size_t)EE * 416 * 2);   // big arena (also eu f32)
  h16* Bbuf   = (h16*)carve((size_t)EE * 256 * 2);   // arena (also gate f32)
  float* ef32 = (float*)carve((size_t)EE * 128 * 4);
  float* nfA  = (float*)carve((size_t)NN * 128 * 4);
  float* nfB  = (float*)carve((size_t)NN * 128 * 4);
  float* gateF = (float*)Bbuf;  // alias: gates computed while B arena is free
  float* euF   = (float*)Abuf;  // alias: eu2 output after concat input is dead

  auto blocks = [](size_t total) { return (unsigned)((total + 255) / 256); };
  auto gemm = [&](const h16* A, const h16* Wt, const float* bias, int Kp, int Np,
                  int Nt, h16* oH, float* oF, int act) {
    if ((Np & 63) == 0) {
      dim3 g(EE / 256, (unsigned)(Np / 64));
      gemm_f16_wmma<4><<<g, 256, 0, stream>>>(A, Wt, bias, EE, Kp, Np, Nt, oH, oF, act);
    } else if (Np == 32) {
      dim3 g(EE / 256, 1);
      gemm_f16_wmma<2><<<g, 256, 0, stream>>>(A, Wt, bias, EE, Kp, Np, Nt, oH, oF, act);
    } else {  // Np == 16
      dim3 g(EE / 256, 1);
      gemm_f16_wmma<1><<<g, 256, 0, stream>>>(A, Wt, bias, EE, Kp, Np, Nt, oH, oF, act);
    }
  };

  // ---- graph construction + geometry ----
  knn_kernel<<<NN / 256, 256, 0, stream>>>(X, mask, src, dst);
  bb5_kernel<<<NN / 256, 256, 0, stream>>>(bb, bb5);

  // ---- weight conversion to fragment-order f16 (with K/N padding) ----
  convert_w_kernel<<<blocks(416 * 256), 256, 0, stream>>>(We1, hWe1, 416, 256, 416, 256);
  convert_w_kernel<<<blocks(256 * 256), 256, 0, stream>>>(We2, hWe2, 256, 256, 256, 256);
  convert_w_kernel<<<blocks(256 * 128), 256, 0, stream>>>(We3, hWe3, 256, 128, 256, 128);
  convert_w_kernel<<<blocks(128 * 32), 256, 0, stream>>>(Wg, hWg, 128, 32, 128, 32);
  convert_w_kernel<<<blocks(192 * 16), 256, 0, stream>>>(Wg0, hWg0, 192, 16, 192, 16);
  convert_w_kernel<<<blocks(160 * 256), 256, 0, stream>>>(Weu1_0, hWeu10, 160, 256, 160, 256);
  convert_w_kernel<<<blocks(256 * 128), 256, 0, stream>>>(Weu2_0, hWeu20, 256, 128, 256, 128);
  convert_w_kernel<<<blocks(160 * 16), 256, 0, stream>>>(Wg1, hWg1, 160, 8, 160, 16);

  // ---- edge features + edge MLP (WMMA, all hops fragment-order) ----
  edge_feat_kernel<<<blocks((size_t)EE * 416 / 16), 256, 0, stream>>>(bb5, src, dst, Abuf);
  gemm(Abuf, hWe1, be1, 416, 256, 256, Bbuf, nullptr, 1);            // relu
  gemm(Bbuf, hWe2, be2, 256, 256, 256, Abuf, nullptr, 1);            // relu
  gemm(Abuf, hWe3, be3, 256, 128, 128, nullptr, ef32, 0);
  layernorm_kernel<<<EE / 8, 256, 0, stream>>>(ef32, nullptr, ge, bge, ef32, EE);

  // ---- initial gating + segment sum ----
  concat_kernel<<<blocks((size_t)EE * 128 / 16), 256, 0, stream>>>(
      nullptr, 0, src, dst, ef32, Abuf, 128);
  gemm(Abuf, hWg, bg, 128, 32, 32, nullptr, gateF, 2);               // sigmoid
  fill0_kernel<<<blocks((size_t)NN * 128), 256, 0, stream>>>(nfA, NN * 128);
  scatter0_kernel<<<blocks((size_t)EE * 128), 256, 0, stream>>>(node, gateF, src, dst, nfA);
  mask_mul_kernel<<<blocks((size_t)NN * 128), 256, 0, stream>>>(nfA, mask, 128);

  // ---- layer 0: ci=32 -> co=16 ----
  concat_kernel<<<blocks((size_t)EE * 192 / 16), 256, 0, stream>>>(nfA, 32, src, dst, ef32, Abuf, 192);
  gemm(Abuf, hWg0, bg0, 192, 16, 16, nullptr, gateF, 2);
  fill0_kernel<<<blocks((size_t)NN * 64), 256, 0, stream>>>(nfB, NN * 64);
  scatter_layer_kernel<<<blocks((size_t)EE * 4 * 16), 256, 0, stream>>>(
      nfA, 32, Wv0, 16, gateF, src, dst, nfB);
  mask_mul_kernel<<<blocks((size_t)NN * 64), 256, 0, stream>>>(nfB, mask, 64);
  concat_kernel<<<blocks((size_t)EE * 160 / 16), 256, 0, stream>>>(nfB, 16, src, dst, ef32, Abuf, 160);
  gemm(Abuf, hWeu10, beu1_0, 160, 256, 256, Bbuf, nullptr, 1);       // relu
  gemm(Bbuf, hWeu20, beu2_0, 256, 128, 128, nullptr, euF, 0);
  layernorm_kernel<<<EE / 8, 256, 0, stream>>>(ef32, euF, geu0, bgeu0, ef32, EE);

  // ---- layer 1: ci=16 -> co=8 (its eu/LN edge update is dead code: outputs
  //      depend only on nf, so those two GEMMs are skipped) ----
  concat_kernel<<<blocks((size_t)EE * 160 / 16), 256, 0, stream>>>(nfB, 16, src, dst, ef32, Abuf, 160);
  gemm(Abuf, hWg1, bg1, 160, 16, 8, nullptr, gateF, 2);
  fill0_kernel<<<blocks((size_t)NN * 32), 256, 0, stream>>>(nfA, NN * 32);
  scatter_layer_kernel<<<blocks((size_t)EE * 4 * 8), 256, 0, stream>>>(
      nfB, 16, Wv1, 8, gateF, src, dst, nfA);
  mask_mul_kernel<<<blocks((size_t)NN * 32), 256, 0, stream>>>(nfA, mask, 32);

  // ---- final heads ----
  final_kernel<<<NN / 256, 256, 0, stream>>>(nfA, rot, mask, normg, Wf, bf, Wmu,
                                             bmu, Wlv, blv, (float*)d_out);
  (void)in_sizes; (void)n_in; (void)out_size; (void)ws_size;
}
